// MultiHeadSelfAttention_31233002177157
// MI455X (gfx1250) — matrix operand order repaired and checked by reading, not compiled, not run on hardware
//
#include <hip/hip_runtime.h>
#include <hip/hip_bf16.h>

// ---------------------------------------------------------------------------
// MHA for [2, 2048, 1024], 16 heads x 64. bf16 WMMA with f32 accumulation.
// Tile staging via the Tensor Data Mover (tensor_load_to_lds / TENSORcnt)
// where the tile is a plain strided 2D copy; vectorized gather + LDS scatter
// where a transpose is required (W^T, V^T).
// ---------------------------------------------------------------------------

typedef __attribute__((ext_vector_type(16))) __bf16 bf16x16;
typedef __attribute__((ext_vector_type(8)))  float  f32x8;
typedef __attribute__((ext_vector_type(4)))  unsigned int u32x4;
typedef __attribute__((ext_vector_type(8)))  int  i32x8;
typedef __attribute__((ext_vector_type(4)))  int  i32x4;

#define BATCH   2
#define SEQ     2048
#define DMODEL  1024
#define HEADS   16
#define HDIM    64
#define N3      3072   // 3 * DMODEL

#if defined(__has_builtin)
#  if __has_builtin(__builtin_amdgcn_tensor_load_to_lds)
#    define USE_TDM 1
#  endif
#endif
#ifndef USE_TDM
#  define USE_TDM 0
#endif

__device__ __forceinline__ unsigned short f2bf(float f) {
  unsigned int u = __float_as_uint(f);
  u += 0x7FFFu + ((u >> 16) & 1u);      // round-to-nearest-even
  return (unsigned short)(u >> 16);
}

union FragBits { uint4 u[2]; bf16x16 v; };
union Chunk8   { uint4 q; unsigned short s[8]; };

// Load a 16-element bf16 fragment as two 16-byte LDS reads.
__device__ __forceinline__ bf16x16 load_frag2(const unsigned short* p0,
                                              const unsigned short* p1) {
  FragBits f;
  f.u[0] = *(const uint4*)p0;
  f.u[1] = *(const uint4*)p1;
  return f.v;
}

__device__ __forceinline__ f32x8 zero8() {
  f32x8 z;
#pragma unroll
  for (int r = 0; r < 8; ++r) z[r] = 0.0f;
  return z;
}

#if USE_TDM
// Issue a TDM load of a 2D bf16 tile (tile_d1 rows x tile_d0 elems, row
// stride stride0 elems) from global memory into LDS at byte offset lds_off.
// D# layout per CDNA5 ISA ch.8 (group0 128b, group1 256b; groups 2/3 zero
// => tensors up to 2D, no iteration/padding/multicast).
// clang-23 builtin arity: (v4u g0, v8i g1, v4i g2, v4i g3, v8i, i32 cpol).
__device__ __forceinline__ void tdm_load_2d(const void* gaddr,
                                            unsigned int lds_off,
                                            unsigned int tile_d0,
                                            unsigned int tile_d1,
                                            unsigned long long stride0) {
  unsigned long long ga = (unsigned long long)(size_t)gaddr;
  u32x4 g0;
  g0[0] = 1u;                                    // count=1, user descriptor
  g0[1] = lds_off;                               // lds_addr (bytes)
  g0[2] = (unsigned int)(ga & 0xFFFFFFFFu);      // global_addr[31:0]
  g0[3] = (unsigned int)((ga >> 32) & 0x01FFFFFFu) | (2u << 30); // [56:32], type=2
  i32x8 g1;
  g1[0] = (int)(1u << 16);                       // data_size=1 (2 bytes)
  g1[1] = (int)((tile_d0 & 0xFFFFu) << 16);      // tensor_dim0[15:0] (== tile)
  g1[2] = (int)((tile_d0 >> 16) | ((tile_d1 & 0xFFFFu) << 16)); // dim0 hi / dim1 lo
  g1[3] = (int)((tile_d1 >> 16) | (tile_d0 << 16));             // dim1 hi / tile_dim0
  g1[4] = (int)(tile_d1 & 0xFFFFu);              // tile_dim1 (tile_dim2 = 0)
  g1[5] = (int)(unsigned int)(stride0 & 0xFFFFFFFFu);           // dim0_stride lo32
  g1[6] = (int)(unsigned int)((stride0 >> 32) & 0xFFFFu);       // dim0_stride hi16
  g1[7] = 0;
  i32x4 z4;
  z4[0] = 0; z4[1] = 0; z4[2] = 0; z4[3] = 0;
  i32x8 z8;
#pragma unroll
  for (int i = 0; i < 8; ++i) z8[i] = 0;
  __builtin_amdgcn_tensor_load_to_lds(g0, g1, z4, z4, z8, 0);
}
#endif

__device__ __forceinline__ unsigned int lds_offset_of(const void* p) {
  return (unsigned int)(size_t)p;   // low 32 bits of a flat LDS address = LDS offset
}

// ---------------------------------------------------------------------------
// fp32 -> bf16 conversion (vectorized, grid-stride)
// ---------------------------------------------------------------------------
__global__ void cvt_bf16x4(const float4* __restrict__ src,
                           ushort4* __restrict__ dst, int n4) {
  int i = blockIdx.x * blockDim.x + threadIdx.x;
  int stride = gridDim.x * blockDim.x;
  for (; i < n4; i += stride) {
    float4 f = src[i];
    ushort4 o;
    o.x = f2bf(f.x); o.y = f2bf(f.y); o.z = f2bf(f.z); o.w = f2bf(f.w);
    dst[i] = o;
  }
}

// ---------------------------------------------------------------------------
// QKV projection: [4096,1024]bf16 x [1024,3072]bf16 + bias -> bf16 QKV
// stored head-major: qkv[(which*2+b)*16+h][n][d], Q pre-scaled by 1/sqrt(64).
// Block: 256 threads = 8 waves (2x4), block tile 128x128, wave tile 64x32.
// A tile staged by the TDM; W tile: vectorized gather then LDS transpose.
// ---------------------------------------------------------------------------
__global__ __launch_bounds__(256, 2)
void qkv_gemm_kernel(const unsigned short* __restrict__ A,   // tokens bf16 [4096][1024]
                     const unsigned short* __restrict__ W,   // w_qkv  bf16 [1024][3072]
                     const float* __restrict__ bias,         // [3072]
                     unsigned short* __restrict__ qkv)       // [3][2][16][2048][64]
{
  __shared__ __align__(16) unsigned short sA[128 * 32];   // [row][k]
  __shared__ __align__(16) unsigned short sBt[128 * 32];  // [col][k]  (B transposed)

  const int t     = threadIdx.x;
  const int wave  = t >> 5;
  const int lane  = t & 31;
  const int lhalf = lane >> 4;     // 0/1
  const int lc    = lane & 15;
  const int waveM = wave >> 2;     // 0..1 -> 64 rows each
  const int waveN = wave & 3;      // 0..3 -> 32 cols each
  const int m0    = blockIdx.x * 128;
  const int n0    = blockIdx.y * 128;

  f32x8 acc[4][2];
#pragma unroll
  for (int mf = 0; mf < 4; ++mf)
#pragma unroll
    for (int nf = 0; nf < 2; ++nf) acc[mf][nf] = zero8();

  for (int k0 = 0; k0 < DMODEL; k0 += 32) {
    __syncthreads();
#if USE_TDM
    // A tile: 128 rows x 32 cols, row stride 1024 elems -> one TDM descriptor.
    if (wave == 0) {
      tdm_load_2d(&A[(size_t)m0 * DMODEL + k0], lds_offset_of(sA),
                  /*tile_d0=*/32, /*tile_d1=*/128, /*stride0=*/DMODEL);
    }
#else
#pragma unroll
    for (int i = 0; i < 2; ++i) {
      int idx = t + i * 256;                 // 0..511
      int row = idx >> 2;                    // 0..127
      int ch  = idx & 3;                     // 4 chunks of 8 bf16 per row
      *(uint4*)&sA[row * 32 + ch * 8] =
          *(const uint4*)&A[(size_t)(m0 + row) * DMODEL + k0 + ch * 8];
    }
#endif
    // B tile transposed into LDS: sBt[n][k] = W[k0+k][n0+n].
    // Phase 1: coalesced b128 gather into registers; phase 2: LDS scatter.
    Chunk8 wbuf[2];
#pragma unroll
    for (int i = 0; i < 2; ++i) {
      int c  = t + i * 256;                  // 0..511 chunks of 8 elems
      int kk = c >> 4;                       // 0..31  (row of W tile)
      int n8 = (c & 15) * 8;                 // 0..120 (col offset, lane-consecutive)
      wbuf[i].q = *(const uint4*)&W[(size_t)(k0 + kk) * N3 + n0 + n8];
    }
#pragma unroll
    for (int i = 0; i < 2; ++i) {
      int c  = t + i * 256;
      int kk = c >> 4;
      int n8 = (c & 15) * 8;
#pragma unroll
      for (int j = 0; j < 8; ++j)
        sBt[(n8 + j) * 32 + kk] = wbuf[i].s[j];
    }
#if USE_TDM
    if (wave == 0) __builtin_amdgcn_s_wait_tensorcnt(0);
#endif
    __syncthreads();

    bf16x16 bfrag[2];
#pragma unroll
    for (int nf = 0; nf < 2; ++nf) {
      int col = waveN * 32 + nf * 16 + lc;
      const unsigned short* bp = &sBt[col * 32 + lhalf * 16];
      bfrag[nf] = load_frag2(bp, bp + 8);    // 16 consecutive K values
    }
#pragma unroll
    for (int mf = 0; mf < 4; ++mf) {
      int row = waveM * 64 + mf * 16 + lc;
      const unsigned short* ap = &sA[row * 32 + lhalf * 16];
      bf16x16 afrag = load_frag2(ap, ap + 8);    // 16 consecutive K values
#pragma unroll
      for (int nf = 0; nf < 2; ++nf) {
        acc[mf][nf] = __builtin_amdgcn_wmma_f32_16x16x32_bf16(
            false, afrag, false, bfrag[nf], (short)0, acc[mf][nf], false, false);
      }
    }
  }

  // Epilogue: + bias, scale Q by 1/sqrt(64), scatter to head-major bf16.
#pragma unroll
  for (int nf = 0; nf < 2; ++nf) {
    int j     = n0 + waveN * 32 + nf * 16 + lc;
    float bj  = bias[j];
    int which = j >> 10;            // 0=Q 1=K 2=V
    int rem   = j & 1023;
    int h     = rem >> 6;
    int d     = rem & 63;
    float scale = (which == 0) ? 0.125f : 1.0f;
#pragma unroll
    for (int mf = 0; mf < 4; ++mf) {
      int mbase = m0 + waveM * 64 + mf * 16 + lhalf * 8;
#pragma unroll
      for (int r = 0; r < 8; ++r) {
        int m  = mbase + r;
        int bb = m >> 11;            // batch
        int n  = m & 2047;           // seq pos
        float v = (acc[mf][nf][r] + bj) * scale;
        qkv[((((size_t)which * BATCH + bb) * HEADS + h) * SEQ + n) * HDIM + d] = f2bf(v);
      }
    }
  }
}

// ---------------------------------------------------------------------------
// Flash attention: one block per (b,h) x 64-query tile. 128 threads = 4 waves,
// each wave owns 16 query rows. Online softmax in f32, both GEMMs on bf16 WMMA.
// Q/K tiles staged by the TDM; V: vectorized gather then LDS transpose.
// ---------------------------------------------------------------------------
__global__ __launch_bounds__(128, 2)
void attn_kernel(const unsigned short* __restrict__ qkv,  // [3][2][16][2048][64] bf16
                 float* __restrict__ out)                 // [2][2048][1024] f32
{
  __shared__ __align__(16) unsigned short sQ [64 * 64];   // [q][d]
  __shared__ __align__(16) unsigned short sK [64 * 64];   // [kv][d]  == B for Q*K^T
  __shared__ __align__(16) unsigned short sVt[64 * 64];   // [d][kv]  == B for P*V
  __shared__ __align__(16) unsigned short sP [4][16 * 64];// per-wave P tile (bf16)

  const int t     = threadIdx.x;
  const int wave  = t >> 5;
  const int lane  = t & 31;
  const int lhalf = lane >> 4;
  const int lc    = lane & 15;
  const int q0    = blockIdx.x * 64;
  const int bh    = blockIdx.y;          // b*16 + h
  const int b     = bh >> 4;
  const int h     = bh & 15;

  const size_t headStride = (size_t)SEQ * HDIM;
  const unsigned short* Qg = qkv + (size_t)(0 * 32 + bh) * headStride;
  const unsigned short* Kg = qkv + (size_t)(1 * 32 + bh) * headStride;
  const unsigned short* Vg = qkv + (size_t)(2 * 32 + bh) * headStride;

  // Stage the 64x64 Q tile and load per-wave A-fragments (kept in VGPRs).
#if USE_TDM
  if (wave == 0) {
    tdm_load_2d(&Qg[(size_t)q0 * HDIM], lds_offset_of(sQ), 64, 64, HDIM);
    __builtin_amdgcn_s_wait_tensorcnt(0);
  }
#else
#pragma unroll
  for (int i = 0; i < 4; ++i) {
    int c = t + i * 128;                 // 512 chunks of 8 bf16
    int row = c >> 3, ch = c & 7;
    *(uint4*)&sQ[row * 64 + ch * 8] =
        *(const uint4*)&Qg[(size_t)(q0 + row) * HDIM + ch * 8];
  }
#endif
  __syncthreads();
  bf16x16 qfrag[2];
#pragma unroll
  for (int ks = 0; ks < 2; ++ks) {
    int row = wave * 16 + lc;
    const unsigned short* p = &sQ[row * 64 + ks * 32 + lhalf * 16];
    qfrag[ks] = load_frag2(p, p + 8);
  }

  float mrow[8], lrow[8];
  f32x8 oacc[4];
#pragma unroll
  for (int r = 0; r < 8; ++r) { mrow[r] = -1e30f; lrow[r] = 0.0f; }
#pragma unroll
  for (int df = 0; df < 4; ++df) oacc[df] = zero8();

  for (int j0 = 0; j0 < SEQ; j0 += 64) {
    __syncthreads();
#if USE_TDM
    // K tile: plain 64x64 strided copy -> TDM.
    if (wave == 0) {
      tdm_load_2d(&Kg[(size_t)j0 * HDIM], lds_offset_of(sK), 64, 64, HDIM);
    }
#else
#pragma unroll
    for (int i = 0; i < 4; ++i) {
      int c = t + i * 128;
      int row = c >> 3, ch = c & 7;
      *(uint4*)&sK[row * 64 + ch * 8] =
          *(const uint4*)&Kg[(size_t)(j0 + row) * HDIM + ch * 8];
    }
#endif
    // V tile transposed: sVt[d][kv].
    // Phase 1: coalesced b128 gather into registers; phase 2: LDS scatter.
    Chunk8 vbuf[4];
#pragma unroll
    for (int i = 0; i < 4; ++i) {
      int c  = t + i * 128;                  // 0..511 chunks of 8 elems
      int kv = c >> 3;                       // 0..63
      int d8 = (c & 7) * 8;                  // 0..56 (lane-consecutive)
      vbuf[i].q = *(const uint4*)&Vg[(size_t)(j0 + kv) * HDIM + d8];
    }
#pragma unroll
    for (int i = 0; i < 4; ++i) {
      int c  = t + i * 128;
      int kv = c >> 3;
      int d8 = (c & 7) * 8;
#pragma unroll
      for (int j = 0; j < 8; ++j)
        sVt[(d8 + j) * 64 + kv] = vbuf[i].s[j];
    }
    // Prefetch next iteration's K/V tiles into cache (global_prefetch_b8).
    if (j0 + 64 < SEQ) {
      __builtin_prefetch(&Kg[(size_t)(j0 + 64) * HDIM + t * 64], 0, 3);
      __builtin_prefetch(&Vg[(size_t)(j0 + 64) * HDIM + t * 64], 0, 3);
    }
#if USE_TDM
    if (wave == 0) __builtin_amdgcn_s_wait_tensorcnt(0);
#endif
    __syncthreads();

    // S = Q * K^T  (16 x 64 per wave), Q already carries the 1/sqrt(64) scale
    f32x8 sacc[4];
#pragma unroll
    for (int cf = 0; cf < 4; ++cf) {
      int col = cf * 16 + lc;                 // kv column
      const unsigned short* p0 = &sK[col * 64 + 0  + lhalf * 16];
      const unsigned short* p1 = &sK[col * 64 + 32 + lhalf * 16];
      bf16x16 b0 = load_frag2(p0, p0 + 8);
      bf16x16 b1 = load_frag2(p1, p1 + 8);
      f32x8 z = zero8();
      z = __builtin_amdgcn_wmma_f32_16x16x32_bf16(false, qfrag[0], false, b0,
                                                  (short)0, z, false, false);
      z = __builtin_amdgcn_wmma_f32_16x16x32_bf16(false, qfrag[1], false, b1,
                                                  (short)0, z, false, false);
      sacc[cf] = z;
    }

    // Online softmax. Row r of this half-wave = wave*16 + lhalf*8 + r;
    // its 64 columns live across the 16 lanes of the half x 4 col-fragments.
    float mnew[8];
#pragma unroll
    for (int r = 0; r < 8; ++r) {
      float mx = fmaxf(fmaxf(sacc[0][r], sacc[1][r]),
                       fmaxf(sacc[2][r], sacc[3][r]));
      mx = fmaxf(mx, __shfl_xor(mx, 1));
      mx = fmaxf(mx, __shfl_xor(mx, 2));
      mx = fmaxf(mx, __shfl_xor(mx, 4));
      mx = fmaxf(mx, __shfl_xor(mx, 8));
      mnew[r] = fmaxf(mrow[r], mx);
    }
#pragma unroll
    for (int r = 0; r < 8; ++r) {
      float sum = 0.0f;
#pragma unroll
      for (int cf = 0; cf < 4; ++cf) {
        float p = __expf(sacc[cf][r] - mnew[r]);
        sum += p;
        sP[wave][(lhalf * 8 + r) * 64 + cf * 16 + lc] = f2bf(p);
      }
      sum += __shfl_xor(sum, 1);
      sum += __shfl_xor(sum, 2);
      sum += __shfl_xor(sum, 4);
      sum += __shfl_xor(sum, 8);
      float alpha = __expf(mrow[r] - mnew[r]);
      lrow[r] = lrow[r] * alpha + sum;
      mrow[r] = mnew[r];
#pragma unroll
      for (int df = 0; df < 4; ++df) oacc[df][r] *= alpha;
    }

    // Same-wave LDS store->load round trip for the P re-layout (C -> A frag).
    asm volatile("s_wait_dscnt 0x0" ::: "memory");

    bf16x16 pfrag[2];
#pragma unroll
    for (int ks = 0; ks < 2; ++ks) {
      const unsigned short* p = &sP[wave][lc * 64 + ks * 32 + lhalf * 16];
      pfrag[ks] = load_frag2(p, p + 8);
    }
    // O += P * V
#pragma unroll
    for (int df = 0; df < 4; ++df) {
      int col = df * 16 + lc;                 // output d column
      const unsigned short* p0 = &sVt[col * 64 + 0  + lhalf * 16];
      const unsigned short* p1 = &sVt[col * 64 + 32 + lhalf * 16];
      bf16x16 v0 = load_frag2(p0, p0 + 8);
      bf16x16 v1 = load_frag2(p1, p1 + 8);
      oacc[df] = __builtin_amdgcn_wmma_f32_16x16x32_bf16(false, pfrag[0], false, v0,
                                                         (short)0, oacc[df], false, false);
      oacc[df] = __builtin_amdgcn_wmma_f32_16x16x32_bf16(false, pfrag[1], false, v1,
                                                         (short)0, oacc[df], false, false);
    }
  }

  // Normalize and write [b][n][h*64+d] in f32
#pragma unroll
  for (int df = 0; df < 4; ++df) {
    int d = df * 16 + lc;
#pragma unroll
    for (int r = 0; r < 8; ++r) {
      int n = q0 + wave * 16 + lhalf * 8 + r;
      out[((size_t)(b * SEQ + n)) * DMODEL + h * HDIM + d] = oacc[df][r] / lrow[r];
    }
  }
}

// ---------------------------------------------------------------------------
extern "C" void kernel_launch(void* const* d_in, const int* in_sizes, int n_in,
                              void* d_out, int out_size, void* d_ws, size_t ws_size,
                              hipStream_t stream) {
  const float* tokens = (const float*)d_in[0];   // [2,2048,1024]
  const float* w_qkv  = (const float*)d_in[1];   // [1024,3072]
  const float* b_qkv  = (const float*)d_in[2];   // [3072]
  float* out = (float*)d_out;

  // Workspace partition (bf16 shadows + head-major QKV), ~40 MB total.
  unsigned short* tok16 = (unsigned short*)d_ws;
  unsigned short* w16   = tok16 + (size_t)BATCH * SEQ * DMODEL;
  unsigned short* qkv16 = w16 + (size_t)DMODEL * N3;

  {
    int n4 = (BATCH * SEQ * DMODEL) / 4;
    cvt_bf16x4<<<dim3(1024), dim3(256), 0, stream>>>((const float4*)tokens,
                                                     (ushort4*)tok16, n4);
  }
  {
    int n4 = (DMODEL * N3) / 4;
    cvt_bf16x4<<<dim3(768), dim3(256), 0, stream>>>((const float4*)w_qkv,
                                                    (ushort4*)w16, n4);
  }
  // 4096/128 x 3072/128 tiles
  qkv_gemm_kernel<<<dim3(32, 24), dim3(256), 0, stream>>>(tok16, w16, b_qkv, qkv16);
  // 2048/64 query tiles x 32 (b,h) pairs
  attn_kernel<<<dim3(32, 32), dim3(128), 0, stream>>>(qkv16, out);
}
